// DINOv3Adapter_25640954757662
// MI455X (gfx1250) — matrix operand order repaired and checked by reading, not compiled, not run on hardware
//
#include <hip/hip_runtime.h>
#include <hip/hip_bf16.h>
#include <math.h>

// ---- problem constants (match reference) ----
#define Bn   4
#define Dm   1024
#define Hh   16
#define Pp   4
#define LCn  5376          // 21*256 query tokens
#define LXn  1024          // 32*32 feat tokens
#define DHn  64
#define HIDn 256
#define HTn  32
#define WTn  32
#define EPSf 1e-6f

typedef __bf16 bf16_t;
typedef __attribute__((ext_vector_type(16))) __bf16 bf16x16;
typedef __attribute__((ext_vector_type(8)))  float  f32x8;
typedef __attribute__((ext_vector_type(8)))  unsigned int u32x8;

// Optional CDNA5 async global->LDS path (ASYNCcnt-tracked, ISA ch.08).
// Probe round 2 leaked the builtin's signature:
//   (int __vector(4) addrspace(1)*, int __vector(4) addrspace(3)*, imm int, imm int)
#if __has_builtin(__builtin_amdgcn_global_load_async_to_lds_b128) && \
    __has_builtin(__builtin_amdgcn_s_wait_asynccnt)
#define HAVE_ASYNC_LDS 1
typedef int v4i_t __attribute__((vector_size(16)));
typedef __attribute__((address_space(1))) v4i_t gv4i_t;
typedef __attribute__((address_space(3))) v4i_t lv4i_t;
#else
#define HAVE_ASYNC_LDS 0
#endif

static __device__ __forceinline__ bf16_t f2bf(float f) {
  unsigned int u = __builtin_bit_cast(unsigned int, f);
  u += 0x7fffu + ((u >> 16) & 1u);                 // round-to-nearest-even
  unsigned short h = (unsigned short)(u >> 16);
  return __builtin_bit_cast(bf16_t, h);
}

// ---------------------------------------------------------------------------
// Pack fp32 weight [K,N] -> bf16 transposed [N,K] (so WMMA B-frags load as
// contiguous bf16 pairs per dword).
// ---------------------------------------------------------------------------
__global__ void pack_wT_kernel(const float* __restrict__ W, bf16_t* __restrict__ Wt,
                               int K, int N) {
  int i = blockIdx.x * 256 + threadIdx.x;
  if (i >= K * N) return;
  int k = i / N, n = i - k * N;
  Wt[(size_t)n * K + k] = f2bf(W[i]);
}

// ---------------------------------------------------------------------------
// LayerNorm rows of fp32 [rows, D] -> bf16 out. One block (256 thr) per row.
// ---------------------------------------------------------------------------
__global__ void layernorm_bf16_kernel(const float* __restrict__ x,
                                      const float* __restrict__ g,
                                      const float* __restrict__ b,
                                      bf16_t* __restrict__ out, int D) {
  const int row = blockIdx.x;
  const float* xr = x + (size_t)row * D;
  float s = 0.f, s2 = 0.f;
  for (int i = threadIdx.x; i < D; i += 256) {
    float v = xr[i];
    s += v; s2 += v * v;
  }
  __shared__ float ssum[256], ssq[256];
  ssum[threadIdx.x] = s; ssq[threadIdx.x] = s2;
  __syncthreads();
  for (int st = 128; st > 0; st >>= 1) {
    if (threadIdx.x < st) {
      ssum[threadIdx.x] += ssum[threadIdx.x + st];
      ssq[threadIdx.x]  += ssq[threadIdx.x + st];
    }
    __syncthreads();
  }
  const float mean = ssum[0] / (float)D;
  const float var  = ssq[0] / (float)D - mean * mean;
  const float rstd = rsqrtf(var + EPSf);
  bf16_t* orow = out + (size_t)row * D;
  for (int i = threadIdx.x; i < D; i += 256)
    orow[i] = f2bf((xr[i] - mean) * rstd * g[i] + b[i]);
}

// ---------------------------------------------------------------------------
// WMMA bf16 GEMM:  C[M,N] = A[M,K](bf16) * Bt[N,K]^T(bf16) + bias + residual
//
// block = 128 threads = 4 waves in a 2x2 wave grid; block tile 64(M) x 64(N);
// K staged through LDS in chunks of 64 (coalesced 16B/lane copies, async
// global->LDS when available).  Each wave owns a 32x32 register tile
// (2x2 accumulators), so each staged chunk feeds 8 v_wmma per wave.
//
// A frag (16x32, 16-bit): dword d, half=lane>=16: k = 2*(d&3) + 16*(d>>2) + 8*half
// B frag (32x16, 16-bit): dword d:                k = 2*(d&3) + 16*(d>>2) + 8*half
// C/D: vgpr r -> row = r + 8*half, col = lane&15
// ---------------------------------------------------------------------------
#define KC  64     // K elements staged per iteration
#define LDT 72     // LDS row stride in bf16 elements (144B, 16B aligned, pads banks)

__global__ void gemm_bf16_wmma_kernel(const bf16_t* __restrict__ A,
                                      const bf16_t* __restrict__ Bt,
                                      const float* __restrict__ bias,
                                      const float* __restrict__ residual,
                                      float* __restrict__ C,
                                      int M, int N, int K) {
  __shared__ __attribute__((aligned(16))) bf16_t As[64 * LDT];
  __shared__ __attribute__((aligned(16))) bf16_t Bs[64 * LDT];

  const int tid  = threadIdx.x;
  const int lane = tid & 31;
  const int wv   = tid >> 5;
  const int wr   = wv >> 1;          // wave row (0..1)
  const int wc   = wv & 1;           // wave col (0..1)
  const int half = lane >> 4;
  const int l15  = lane & 15;
  const int m0   = blockIdx.x * 64;
  const int n0   = blockIdx.y * 64;

  f32x8 acc[2][2] = {};

  for (int k0 = 0; k0 < K; k0 += KC) {
    // ---- stage 64x64-element A and B tiles into LDS (coalesced) ----
#pragma unroll
    for (int i = 0; i < 4; ++i) {
      const int c   = tid + 128 * i;       // 512 chunks of 16B per tile
      const int row = c >> 3;              // 0..63
      const int ce  = (c & 7) * 8;         // bf16 element offset within row
      const bf16_t* ga = A  + (size_t)(m0 + row) * K + k0 + ce;
      const bf16_t* gb = Bt + (size_t)(n0 + row) * K + k0 + ce;
      bf16_t* la = &As[row * LDT + ce];
      bf16_t* lb = &Bs[row * LDT + ce];
#if HAVE_ASYNC_LDS
      __builtin_amdgcn_global_load_async_to_lds_b128((gv4i_t*)ga, (lv4i_t*)la, 0, 0);
      __builtin_amdgcn_global_load_async_to_lds_b128((gv4i_t*)gb, (lv4i_t*)lb, 0, 0);
#else
      *(uint4*)la = *(const uint4*)ga;
      *(uint4*)lb = *(const uint4*)gb;
#endif
    }
#if HAVE_ASYNC_LDS
    __builtin_amdgcn_s_wait_asynccnt(0);
#endif
    __syncthreads();

    // ---- two 16x16x32 WMMA K-steps per staged chunk ----
#pragma unroll
    for (int ks = 0; ks < KC; ks += 32) {
      u32x8 afrag[2], bfrag[2];
#pragma unroll
      for (int mi = 0; mi < 2; ++mi) {
        const bf16_t* ar = &As[(wr * 32 + mi * 16 + l15) * LDT + ks];
#pragma unroll
        for (int d = 0; d < 8; ++d) {
          const int ka = ((d & 3) << 1) + ((d >> 2) << 4) + (half << 3);
          afrag[mi][d] = *(const unsigned int*)(ar + ka);
        }
      }
#pragma unroll
      for (int nj = 0; nj < 2; ++nj) {
        const bf16_t* br = &Bs[(wc * 32 + nj * 16 + l15) * LDT + ks];
#pragma unroll
        for (int d = 0; d < 8; ++d) {
          const int kb = ((d & 3) << 1) + ((d >> 2) << 4) + (half << 3);
          bfrag[nj][d] = *(const unsigned int*)(br + kb);
        }
      }
#pragma unroll
      for (int mi = 0; mi < 2; ++mi)
#pragma unroll
        for (int nj = 0; nj < 2; ++nj)
          acc[mi][nj] = __builtin_amdgcn_wmma_f32_16x16x32_bf16(
              false, __builtin_bit_cast(bf16x16, afrag[mi]),
              false, __builtin_bit_cast(bf16x16, bfrag[nj]),
              (short)0, acc[mi][nj], false, false);
    }
    __syncthreads();
  }

  // ---- epilogue: bias + optional residual, fp32 store ----
#pragma unroll
  for (int mi = 0; mi < 2; ++mi) {
#pragma unroll
    for (int nj = 0; nj < 2; ++nj) {
      const int col = n0 + wc * 32 + nj * 16 + l15;
      const float bn = bias ? bias[col] : 0.0f;
#pragma unroll
      for (int r = 0; r < 8; ++r) {
        const int row = m0 + wr * 32 + mi * 16 + r + 8 * half;
        const size_t o = (size_t)row * N + col;
        float v = acc[mi][nj][r] + bn;
        if (residual) v += residual[o];
        C[o] = v;
      }
    }
  }
}

// ---------------------------------------------------------------------------
// Deformable sampler: softmax over P=4 + 4-tap bilinear gather per
// (b, lc, h); 64 lanes cover dh.  Writes attn as bf16 [B*LC, D].
// ---------------------------------------------------------------------------
__global__ void deform_sample_kernel(const float* __restrict__ off,   // [B*LC,128]
                                     const float* __restrict__ awl,   // [B*LC, 64]
                                     const float* __restrict__ value, // [B*LX, D]
                                     const float* __restrict__ refp,  // [LC,2]
                                     bf16_t* __restrict__ attnb) {    // [B*LC, D]
  const int g  = blockIdx.x * 4 + (threadIdx.x >> 6);
  const int dh = threadIdx.x & 63;
  const int h  = g % Hh;
  const int t  = g / Hh;
  const int lc = t % LCn;
  const int b  = t / LCn;
  const size_t rowq = (size_t)b * LCn + lc;

  const float* op = off + rowq * (Hh * Pp * 2) + h * (Pp * 2);
  const float* ap = awl + rowq * (Hh * Pp) + h * Pp;

  // softmax over P
  float a0 = ap[0], a1 = ap[1], a2 = ap[2], a3 = ap[3];
  float mx = fmaxf(fmaxf(a0, a1), fmaxf(a2, a3));
  float e0 = __expf(a0 - mx), e1 = __expf(a1 - mx),
        e2 = __expf(a2 - mx), e3 = __expf(a3 - mx);
  float inv = 1.0f / (e0 + e1 + e2 + e3);
  float w4[4] = {e0 * inv, e1 * inv, e2 * inv, e3 * inv};

  const float rx = refp[lc * 2 + 0];
  const float ry = refp[lc * 2 + 1];
  const float* vbase = value + ((size_t)b * LXn) * Dm + h * DHn + dh;

  float acc = 0.f;
#pragma unroll
  for (int p = 0; p < Pp; ++p) {
    const float x = (rx + op[p * 2 + 0] / (float)WTn) * (float)WTn - 0.5f;
    const float y = (ry + op[p * 2 + 1] / (float)HTn) * (float)HTn - 0.5f;
    const float x0f = floorf(x), y0f = floorf(y);
    const float fx = x - x0f, fy = y - y0f;
    const int ix = (int)x0f, iy = (int)y0f;
    float s = 0.f;
#pragma unroll
    for (int dy = 0; dy < 2; ++dy) {
#pragma unroll
      for (int dx = 0; dx < 2; ++dx) {
        const int xi = ix + dx, yi = iy + dy;
        if (xi >= 0 && xi < WTn && yi >= 0 && yi < HTn) {
          const float w = (dx ? fx : 1.f - fx) * (dy ? fy : 1.f - fy);
          s += w * vbase[(size_t)(yi * WTn + xi) * Dm];
        }
      }
    }
    acc += w4[p] * s;
  }
  attnb[rowq * Dm + h * DHn + dh] = f2bf(acc);
}

// ---------------------------------------------------------------------------
// Depthwise 3x3 'SAME' over the 64x64 / 32x32 / 16x16 pyramid + bias + exact
// GELU.  block = 256 threads (one per channel), one block per (b, token).
// ---------------------------------------------------------------------------
__global__ void dwconv_gelu_kernel(const float* __restrict__ h1,  // [B*LC,HID]
                                   const float* __restrict__ kw,  // [3,3,1,HID]
                                   const float* __restrict__ kb,  // [HID]
                                   bf16_t* __restrict__ out) {    // [B*LC,HID]
  const int ch = threadIdx.x;
  const int t  = blockIdx.x % LCn;
  const int b  = blockIdx.x / LCn;

  int base, H0, W0, y, x;
  if (t < 4096)        { base = 0;    H0 = 64; W0 = 64; int u = t;        y = u >> 6; x = u & 63; }
  else if (t < 5120)   { base = 4096; H0 = 32; W0 = 32; int u = t - 4096; y = u >> 5; x = u & 31; }
  else                 { base = 5120; H0 = 16; W0 = 16; int u = t - 5120; y = u >> 4; x = u & 15; }

  const float* src = h1 + ((size_t)b * LCn + base) * HIDn + ch;
  float acc = kb[ch];
#pragma unroll
  for (int ky = 0; ky < 3; ++ky) {
#pragma unroll
    for (int kx = 0; kx < 3; ++kx) {
      const int yy = y + ky - 1, xx = x + kx - 1;
      if (yy >= 0 && yy < H0 && xx >= 0 && xx < W0)
        acc += src[(size_t)(yy * W0 + xx) * HIDn] * kw[(ky * 3 + kx) * HIDn + ch];
    }
  }
  const float ge = 0.5f * acc * (1.0f + erff(acc * 0.70710678118654752f));
  out[((size_t)b * LCn + t) * HIDn + ch] = f2bf(ge);
}

// ---------------------------------------------------------------------------
extern "C" void kernel_launch(void* const* d_in, const int* in_sizes, int n_in,
                              void* d_out, int out_size, void* d_ws, size_t ws_size,
                              hipStream_t stream) {
  (void)in_sizes; (void)n_in; (void)out_size; (void)ws_size;
  const float* query   = (const float*)d_in[0];
  const float* feat    = (const float*)d_in[1];
  const float* refp    = (const float*)d_in[2];
  const float* ln_q_g  = (const float*)d_in[3];
  const float* ln_q_b  = (const float*)d_in[4];
  const float* ln_f_g  = (const float*)d_in[5];
  const float* ln_f_b  = (const float*)d_in[6];
  const float* W_value = (const float*)d_in[7];
  const float* b_value = (const float*)d_in[8];
  const float* W_off   = (const float*)d_in[9];
  const float* b_off   = (const float*)d_in[10];
  const float* W_attn  = (const float*)d_in[11];
  const float* b_attn  = (const float*)d_in[12];
  const float* W_out   = (const float*)d_in[13];
  const float* b_out   = (const float*)d_in[14];
  const float* ln_ffn_g= (const float*)d_in[15];
  const float* ln_ffn_b= (const float*)d_in[16];
  const float* fc1_W   = (const float*)d_in[17];
  const float* fc1_b   = (const float*)d_in[18];
  const float* dw_W    = (const float*)d_in[19];
  const float* dw_b    = (const float*)d_in[20];
  const float* fc2_W   = (const float*)d_in[21];
  const float* fc2_b   = (const float*)d_in[22];
  float* out = (float*)d_out;

  const int MQ = Bn * LCn;   // 21504
  const int MF = Bn * LXn;   // 4096

  // ---- workspace layout ----
  char* cur = (char*)d_ws;
  auto alloc = [&](size_t bytes) -> char* {
    char* p = cur; cur += (bytes + 255) & ~(size_t)255; return p;
  };
  bf16_t* WvT   = (bf16_t*)alloc((size_t)Dm * Dm * 2);
  bf16_t* WoffT = (bf16_t*)alloc((size_t)Dm * 128 * 2);
  bf16_t* WatT  = (bf16_t*)alloc((size_t)Dm * 64 * 2);
  bf16_t* WoutT = (bf16_t*)alloc((size_t)Dm * Dm * 2);
  bf16_t* Wf1T  = (bf16_t*)alloc((size_t)Dm * HIDn * 2);
  bf16_t* Wf2T  = (bf16_t*)alloc((size_t)HIDn * Dm * 2);
  bf16_t* fnb   = (bf16_t*)alloc((size_t)MF * Dm * 2);
  float*  value = (float*) alloc((size_t)MF * Dm * 4);
  bf16_t* qb    = (bf16_t*)alloc((size_t)MQ * Dm * 2);   // reused: qn, attn, qn2
  float*  offo  = (float*) alloc((size_t)MQ * 128 * 4);
  float*  awlo  = (float*) alloc((size_t)MQ * 64 * 4);
  float*  query2= (float*) alloc((size_t)MQ * Dm * 4);
  float*  h1    = (float*) alloc((size_t)MQ * HIDn * 4);
  bf16_t* h1g   = (bf16_t*)alloc((size_t)MQ * HIDn * 2);

  // ---- 1) pack weights (fp32 [K,N] -> bf16 [N,K]) ----
  auto packN = [&](const float* W, bf16_t* Wt, int K, int N) {
    int tot = K * N;
    pack_wT_kernel<<<(tot + 255) / 256, 256, 0, stream>>>(W, Wt, K, N);
  };
  packN(W_value, WvT,  Dm, Dm);
  packN(W_off,   WoffT,Dm, 128);
  packN(W_attn,  WatT, Dm, 64);
  packN(W_out,   WoutT,Dm, Dm);
  packN(fc1_W,   Wf1T, Dm, HIDn);
  packN(fc2_W,   Wf2T, HIDn, Dm);

  // ---- 2) LN(feat) -> bf16 ; value = fn @ W_value + b_value ----
  layernorm_bf16_kernel<<<MF, 256, 0, stream>>>(feat, ln_f_g, ln_f_b, fnb, Dm);
  gemm_bf16_wmma_kernel<<<dim3(MF / 64, Dm / 64), 128, 0, stream>>>(
      fnb, WvT, b_value, nullptr, value, MF, Dm, Dm);

  // ---- 3) LN(query) -> bf16 ; offsets + attn logits ----
  layernorm_bf16_kernel<<<MQ, 256, 0, stream>>>(query, ln_q_g, ln_q_b, qb, Dm);
  gemm_bf16_wmma_kernel<<<dim3(MQ / 64, 128 / 64), 128, 0, stream>>>(
      qb, WoffT, b_off, nullptr, offo, MQ, 128, Dm);
  gemm_bf16_wmma_kernel<<<dim3(MQ / 64, 64 / 64), 128, 0, stream>>>(
      qb, WatT, b_attn, nullptr, awlo, MQ, 64, Dm);

  // ---- 4) deformable sampling (softmax + bilinear), attn -> bf16 (reuse qb) ----
  deform_sample_kernel<<<(Bn * LCn * Hh) / 4, 256, 0, stream>>>(
      offo, awlo, value, refp, qb);

  // ---- 5) out-proj + residual(query) -> query2 ----
  gemm_bf16_wmma_kernel<<<dim3(MQ / 64, Dm / 64), 128, 0, stream>>>(
      qb, WoutT, b_out, query, query2, MQ, Dm, Dm);

  // ---- 6) ConvFFN: LN -> fc1 -> dwconv+GELU -> fc2 + residual(query2) ----
  layernorm_bf16_kernel<<<MQ, 256, 0, stream>>>(query2, ln_ffn_g, ln_ffn_b, qb, Dm);
  gemm_bf16_wmma_kernel<<<dim3(MQ / 64, HIDn / 64), 128, 0, stream>>>(
      qb, Wf1T, fc1_b, nullptr, h1, MQ, HIDn, Dm);
  dwconv_gelu_kernel<<<Bn * LCn, HIDn, 0, stream>>>(h1, dw_W, dw_b, h1g);
  gemm_bf16_wmma_kernel<<<dim3(MQ / 64, Dm / 64), 128, 0, stream>>>(
      h1g, Wf2T, fc2_b, query2, out, MQ, Dm, HIDn);
}
